// Network_15264313770550
// MI455X (gfx1250) — compile-verified, matrix operand order checked by reading, not run on hardware
//
#include <hip/hip_runtime.h>
#include <hip/hip_bf16.h>

typedef __attribute__((ext_vector_type(16))) _Float16     v16h;
typedef __attribute__((ext_vector_type(8)))  float        v8f;
typedef __attribute__((ext_vector_type(4)))  float        f32x4;
typedef __attribute__((ext_vector_type(4)))  unsigned int u32x4;

#define S_   256
#define B_   512
#define IN_  512
#define HID_ 1024
#define A_   16
#define LH_  256
#define LIN_ 1041
#define NB_  (S_ * B_)   // 131072 rows

// ---------------------------------------------------------------------------
// WMMA fragment load (16-bit A or B matrix, 16x32 / 32x16 tile) from LDS.
// Layout per CDNA5 ISA 7.12.2: lane L holds row/col (L&15); K index for
// vector element e: v=e>>1, h=e&1, K = (lane>=16 ? 8:0) + 2v + h + (v>=4 ? 8:0)
// Compiles to two ds_load_b128 per fragment.
// ---------------------------------------------------------------------------
__device__ __forceinline__ v16h frag_ld(const _Float16* p, int ld) {
  const int lane = threadIdx.x & 31;
  const int r    = lane & 15;
  const int khi  = (lane & 16) ? 8 : 0;
  v16h f;
#pragma unroll
  for (int e = 0; e < 16; ++e) {
    const int v = e >> 1, h = e & 1;
    const int k = khi + (v << 1) + h + ((v >= 4) ? 8 : 0);
    f[e] = p[r * ld + k];
  }
  return f;
}

__device__ __forceinline__ float sigm(float x) { return 1.0f / (1.0f + __expf(-x)); }

// ---------------------------------------------------------------------------
// Utility: f32 -> f16 2D convert (handles strided source rows, e.g. W_ih 1041)
// ---------------------------------------------------------------------------
__global__ void k_cvt2d(const float* __restrict__ src, _Float16* __restrict__ dst,
                        int rows, int cols, int sld) {
  int e = blockIdx.x * 256 + threadIdx.x;
  if (e < rows * cols) {
    int r = e / cols, c = e - r * cols;
    dst[e] = (_Float16)src[(size_t)r * sld + c];
  }
}

__global__ void k_zero(unsigned int* bar) {
  if (threadIdx.x == 0 && blockIdx.x == 0)
    __hip_atomic_store(bar, 0u, __ATOMIC_RELAXED, __HIP_MEMORY_SCOPE_AGENT);
}

// ---------------------------------------------------------------------------
// GEMM1: H1 = leaky_relu(X @ W1^T + b1), f16 out.  Tile: BM=128, BN=128, BK=32.
// 8 waves; wave (wr, wc): 4 M-subtiles x 2 N-subtiles = 8 accumulators,
// 6 fragment loads -> 8 back-to-back WMMAs.  X / H1 are single-use streams:
// non-temporal to keep L2 for the LSTM state.
// ---------------------------------------------------------------------------
__global__ __launch_bounds__(256) void k_gemm1(const float* __restrict__ X,
                                               const _Float16* __restrict__ W1h,
                                               const float* __restrict__ b1,
                                               _Float16* __restrict__ H1) {
  __shared__ _Float16 sA[128 * 32];
  __shared__ _Float16 sB[128 * 32];
  const int tid  = threadIdx.x;
  const int n0   = blockIdx.x * 128;
  const int j0   = blockIdx.y * 128;
  const int wave = tid >> 5, lane = tid & 31;
  const int wr = wave >> 2, wc = wave & 3;

  v8f acc[4][2];
#pragma unroll
  for (int m = 0; m < 4; ++m)
#pragma unroll
    for (int nn = 0; nn < 2; ++nn)
#pragma unroll
      for (int i = 0; i < 8; ++i) acc[m][nn][i] = 0.0f;

  for (int k0 = 0; k0 < IN_; k0 += 32) {
    __syncthreads();
    {
      // sA: 128x32 f32 -> f16; 16 floats per thread (4x b128 NT -> 2x ds b128)
      const int f = tid * 16;               // 4096 floats total
      const int row = f >> 5, col = f & 31; // col in {0,16}
      const f32x4* src = (const f32x4*)(X + (size_t)(n0 + row) * IN_ + k0 + col);
      f32x4 x0 = __builtin_nontemporal_load(src + 0);
      f32x4 x1 = __builtin_nontemporal_load(src + 1);
      f32x4 x2 = __builtin_nontemporal_load(src + 2);
      f32x4 x3 = __builtin_nontemporal_load(src + 3);
      union { _Float16 h[16]; u32x4 q[2]; } u;
#pragma unroll
      for (int i = 0; i < 4; ++i) {
        u.h[i]      = (_Float16)x0[i];
        u.h[4 + i]  = (_Float16)x1[i];
        u.h[8 + i]  = (_Float16)x2[i];
        u.h[12 + i] = (_Float16)x3[i];
      }
      u32x4* d = (u32x4*)&sA[f];
      d[0] = u.q[0]; d[1] = u.q[1];
    }
    {
      // sB: 128x32 f16; 16 halves per thread (2x b128)
      const int f = tid * 16;               // 4096 halves total
      const int row = f >> 5, col = f & 31;
      const u32x4* src = (const u32x4*)(W1h + (size_t)(j0 + row) * IN_ + k0 + col);
      u32x4* d = (u32x4*)&sA[0];            // placeholder, replaced below
      (void)d;
      u32x4* db = (u32x4*)&sB[f];
      db[0] = src[0]; db[1] = src[1];
    }
    __syncthreads();
    v16h af[4], bf[2];
#pragma unroll
    for (int m = 0; m < 4; ++m)  af[m] = frag_ld(&sA[(wr * 64 + m * 16) * 32], 32);
#pragma unroll
    for (int nn = 0; nn < 2; ++nn) bf[nn] = frag_ld(&sB[(wc * 32 + nn * 16) * 32], 32);
#pragma unroll
    for (int m = 0; m < 4; ++m)
#pragma unroll
      for (int nn = 0; nn < 2; ++nn)
        acc[m][nn] = __builtin_amdgcn_wmma_f32_16x16x32_f16(false, af[m], false, bf[nn],
                                                            (short)0, acc[m][nn], false, false);
  }

#pragma unroll
  for (int nn = 0; nn < 2; ++nn) {
    const int   j  = j0 + wc * 32 + nn * 16 + (lane & 15);
    const float bj = b1[j];
#pragma unroll
    for (int m = 0; m < 4; ++m)
#pragma unroll
      for (int i = 0; i < 8; ++i) {
        int   n = n0 + wr * 64 + m * 16 + ((lane & 16) ? (8 + i) : i);
        float v = acc[m][nn][i] + bj;
        v = (v > 0.0f) ? v : 0.01f * v;     // leaky_relu slope 0.01
        __builtin_nontemporal_store((_Float16)v, &H1[(size_t)n * HID_ + j]);
      }
  }
}

// ---------------------------------------------------------------------------
// GEMM2: GX = H1 @ W_ih[:, :1024]^T + b_ih + b_hh
//            + reward * W_ih[:,1024] + W_ih[:, 1025 + action]        (f32 out)
// The last 17 columns of W_ih multiply [reward, one_hot] -> rank-1 + gather.
// ---------------------------------------------------------------------------
__global__ __launch_bounds__(256) void k_gemm2(const _Float16* __restrict__ H1,
                                               const _Float16* __restrict__ Wihh,
                                               const float* __restrict__ Wih32,
                                               const float* __restrict__ b_ih,
                                               const float* __restrict__ b_hh,
                                               const float* __restrict__ reward,
                                               const long long* __restrict__ act,
                                               float* __restrict__ GX) {
  __shared__ _Float16 sA[128 * 32];
  __shared__ _Float16 sB[128 * 32];
  const int tid  = threadIdx.x;
  const int n0   = blockIdx.x * 128;
  const int j0   = blockIdx.y * 128;
  const int wave = tid >> 5, lane = tid & 31;
  const int wr = wave >> 2, wc = wave & 3;

  v8f acc[4][2];
#pragma unroll
  for (int m = 0; m < 4; ++m)
#pragma unroll
    for (int nn = 0; nn < 2; ++nn)
#pragma unroll
      for (int i = 0; i < 8; ++i) acc[m][nn][i] = 0.0f;

  for (int k0 = 0; k0 < HID_; k0 += 32) {
    __syncthreads();
    {
      // sA: 128x32 f16; 16 halves per thread (2x b128, NT: single-use stream)
      const int f = tid * 16;
      const int row = f >> 5, col = f & 31;
      const u32x4* src = (const u32x4*)(H1 + (size_t)(n0 + row) * HID_ + k0 + col);
      u32x4 a0 = __builtin_nontemporal_load(src + 0);
      u32x4 a1 = __builtin_nontemporal_load(src + 1);
      u32x4* d = (u32x4*)&sA[f];
      d[0] = a0; d[1] = a1;
    }
    {
      const int f = tid * 16;
      const int row = f >> 5, col = f & 31;
      const u32x4* src = (const u32x4*)(Wihh + (size_t)(j0 + row) * HID_ + k0 + col);
      u32x4* d = (u32x4*)&sB[f];
      d[0] = src[0]; d[1] = src[1];
    }
    __syncthreads();
    v16h af[4], bf[2];
#pragma unroll
    for (int m = 0; m < 4; ++m)  af[m] = frag_ld(&sA[(wr * 64 + m * 16) * 32], 32);
#pragma unroll
    for (int nn = 0; nn < 2; ++nn) bf[nn] = frag_ld(&sB[(wc * 32 + nn * 16) * 32], 32);
#pragma unroll
    for (int m = 0; m < 4; ++m)
#pragma unroll
      for (int nn = 0; nn < 2; ++nn)
        acc[m][nn] = __builtin_amdgcn_wmma_f32_16x16x32_f16(false, af[m], false, bf[nn],
                                                            (short)0, acc[m][nn], false, false);
  }

#pragma unroll
  for (int nn = 0; nn < 2; ++nn) {
    const int    j   = j0 + wc * 32 + nn * 16 + (lane & 15);
    const float  bj  = b_ih[j] + b_hh[j];
    const float  wrj = Wih32[(size_t)j * LIN_ + HID_];        // reward column
    const float* waj = Wih32 + (size_t)j * LIN_ + HID_ + 1;   // one-hot columns
#pragma unroll
    for (int m = 0; m < 4; ++m)
#pragma unroll
      for (int i = 0; i < 8; ++i) {
        int   n = n0 + wr * 64 + m * 16 + ((lane & 16) ? (8 + i) : i);
        float v = acc[m][nn][i] + bj + reward[n] * wrj + waj[(int)act[n]];
        __builtin_nontemporal_store(v, &GX[(size_t)n * HID_ + j]);
      }
  }
}

// ---------------------------------------------------------------------------
// Persistent LSTM: 16 workgroups x 512 threads (16 waves, wave32).
// WG wg owns hidden units [16*wg, 16*wg+16) => gate cols {g*256 + u} for all
// 4 gates, so i/f/g/o for a given (batch, unit) live in one lane's regs.
// Wave w computes batch-row tiles {2w, 2w+1}. c-state stays in registers.
// One agent-scope atomic barrier per step orders h(t-1) stores vs h(t) loads.
// ---------------------------------------------------------------------------
__global__ __launch_bounds__(512) void k_lstm(const float* __restrict__ GX,
                                              const _Float16* __restrict__ Whhh,
                                              const _Float16* __restrict__ H0,
                                              const float* __restrict__ hx,
                                              const unsigned char* __restrict__ dones,
                                              _Float16* __restrict__ HS,
                                              unsigned int* __restrict__ bar) {
  __shared__ _Float16 sW[64 * 256];   // 32 KB: this WG's 64 gate rows of W_hh
  __shared__ _Float16 sH[512 * 32];   // 32 KB: h tile for one K-step

  const int tid  = threadIdx.x;
  const int wave = tid >> 5, lane = tid & 31;
  const int hc0  = blockIdx.x * 16;
  const int u    = hc0 + (lane & 15);
  const int m0   = wave * 2;

  // preload this WG's W_hh slice: rows r = g*16+ul -> gate col j = g*256+hc0+ul
  for (int q = tid; q < 2048; q += 512) {       // 2048 x b128 = 32 KB
    const int f = q * 8;
    const int r = f >> 8, k = f & 255;
    const int g = r >> 4, ul = r & 15;
    const int jj = g * LH_ + hc0 + ul;
    ((u32x4*)sW)[q] = *(const u32x4*)(Whhh + (size_t)jj * LH_ + k);
  }

  float creg[2][8];
#pragma unroll
  for (int m = 0; m < 2; ++m)
#pragma unroll
    for (int i = 0; i < 8; ++i) {
      int b = (m0 + m) * 16 + ((lane & 16) ? (8 + i) : i);
      creg[m][i] = hx[(size_t)B_ * LH_ + (size_t)b * LH_ + u];   // hx[1] = c0
    }
  __syncthreads();

  for (int t = 0; t < S_; ++t) {
    const _Float16* hprev = (t == 0) ? H0 : (HS + (size_t)(t - 1) * B_ * LH_);
    const unsigned char* dt = dones + (size_t)t * B_;
    const size_t gbase = (size_t)t * B_ * HID_;

    v8f acc[2][4];
#pragma unroll
    for (int m = 0; m < 2; ++m)
#pragma unroll
      for (int i = 0; i < 8; ++i) {
        int b = (m0 + m) * 16 + ((lane & 16) ? (8 + i) : i);
#pragma unroll
        for (int g = 0; g < 4; ++g)
          acc[m][g][i] = GX[gbase + (size_t)b * HID_ + g * LH_ + u];
      }

    // pull next step's gate slab toward L2 while we grind this step
    if (t + 1 < S_) {
      const float* nx = GX + gbase + (size_t)B_ * HID_
                      + (size_t)(m0 * 16 + (lane & 15)) * HID_ + hc0;
      __builtin_prefetch(nx, 0, 1);
      __builtin_prefetch(nx + (size_t)16 * HID_, 0, 1);
    }

    for (int ks = 0; ks < 8; ++ks) {
      __syncthreads();
      {
        // stage h[0..511][ks*32 .. +32): one row per thread, 4x b128 each way
        const int row = tid;
        const u32x4* src = (const u32x4*)(hprev + (size_t)row * LH_ + ks * 32);
        u32x4 a0 = src[0], a1 = src[1], a2 = src[2], a3 = src[3];
        if (dt[row]) {                       // done => h reset for this step
          a0 = (u32x4)0u; a1 = (u32x4)0u; a2 = (u32x4)0u; a3 = (u32x4)0u;
        }
        u32x4* d = (u32x4*)&sH[row * 32];
        d[0] = a0; d[1] = a1; d[2] = a2; d[3] = a3;
      }
      __syncthreads();
      v16h bf[4], af[2];
#pragma unroll
      for (int g = 0; g < 4; ++g) bf[g] = frag_ld(&sW[(g * 16) * 256 + ks * 32], 256);
#pragma unroll
      for (int m = 0; m < 2; ++m) af[m] = frag_ld(&sH[((m0 + m) * 16) * 32], 32);
#pragma unroll
      for (int m = 0; m < 2; ++m)
#pragma unroll
        for (int g = 0; g < 4; ++g)
          acc[m][g] = __builtin_amdgcn_wmma_f32_16x16x32_f16(false, af[m], false, bf[g],
                                                             (short)0, acc[m][g], false, false);
    }

    _Float16* hout = HS + (size_t)t * B_ * LH_;
#pragma unroll
    for (int m = 0; m < 2; ++m)
#pragma unroll
      for (int i = 0; i < 8; ++i) {
        int   b  = (m0 + m) * 16 + ((lane & 16) ? (8 + i) : i);
        float ig = sigm(acc[m][0][i]);
        float fg = sigm(acc[m][1][i]);
        float gg = tanhf(acc[m][2][i]);
        float og = sigm(acc[m][3][i]);
        float cp = dt[b] ? 0.0f : creg[m][i];        // done => reset c
        float c  = fg * cp + ig * gg;
        creg[m][i] = c;
        hout[(size_t)b * LH_ + u] = (_Float16)(og * tanhf(c));
      }

    // ---- grid barrier: h(t) visible to all 16 WGs before step t+1 ----
    __threadfence();
    __syncthreads();
    if (tid == 0) {
      __hip_atomic_fetch_add(bar, 1u, __ATOMIC_ACQ_REL, __HIP_MEMORY_SCOPE_AGENT);
      const unsigned int target = 16u * (unsigned)(t + 1);
      while (__hip_atomic_load(bar, __ATOMIC_ACQUIRE, __HIP_MEMORY_SCOPE_AGENT) < target)
        __builtin_amdgcn_s_sleep(8);
    }
    __syncthreads();
  }
}

// ---------------------------------------------------------------------------
// Heads: logits = hs @ W_pi^T + b_pi (NaN -> 1e-12), values = hs @ W_v^T + b_v.
// 8 contiguous rows per WG staged in LDS (flat b128 copy); 17 outputs per row.
// ---------------------------------------------------------------------------
__global__ __launch_bounds__(160) void k_heads(const _Float16* __restrict__ HS,
                                               const float* __restrict__ Wpi,
                                               const float* __restrict__ bpi,
                                               const float* __restrict__ Wv,
                                               const float* __restrict__ bv,
                                               float* __restrict__ out) {
  __shared__ _Float16 sh[8 * LH_];
  const int tid = threadIdx.x;
  const int r0  = blockIdx.x * 8;
  const u32x4* src = (const u32x4*)(HS + (size_t)r0 * LH_);   // 8 rows contiguous
  for (int q = tid; q < 256; q += 160) ((u32x4*)sh)[q] = src[q];
  __syncthreads();
  if (tid < 136) {
    const int nl = tid / 17, a = tid % 17;
    const float* wrow = (a < 16) ? (Wpi + (size_t)a * LH_) : Wv;
    float s = (a < 16) ? bpi[a] : bv[0];
    for (int k = 0; k < LH_; ++k) s += (float)sh[nl * LH_ + k] * wrow[k];
    const size_t n = (size_t)r0 + nl;
    if (a < 16) {
      if (s != s) s = 1e-12f;                         // NaN guard (logits only)
      out[n * A_ + a] = s;                            // flat == reshape(S,A,B)
    } else {
      out[(size_t)S_ * A_ * B_ + n] = s;              // values
    }
  }
}

// ---------------------------------------------------------------------------
extern "C" void kernel_launch(void* const* d_in, const int* in_sizes, int n_in,
                              void* d_out, int out_size, void* d_ws, size_t ws_size,
                              hipStream_t stream) {
  const float*         x      = (const float*)d_in[0];
  const float*         reward = (const float*)d_in[1];
  const float*         hx     = (const float*)d_in[2];
  const float*         W1     = (const float*)d_in[3];
  const float*         b1     = (const float*)d_in[4];
  const float*         W_ih   = (const float*)d_in[5];
  const float*         b_ih   = (const float*)d_in[6];
  const float*         W_hh   = (const float*)d_in[7];
  const float*         b_hh   = (const float*)d_in[8];
  const float*         W_pi   = (const float*)d_in[9];
  const float*         b_pi   = (const float*)d_in[10];
  const float*         W_v    = (const float*)d_in[11];
  const float*         b_v    = (const float*)d_in[12];
  const long long*     act    = (const long long*)d_in[13];   // int64
  const unsigned char* dones  = (const unsigned char*)d_in[14];
  float* out = (float*)d_out;

  char* w = (char*)d_ws;
  _Float16* W1h  = (_Float16*)w; w += (size_t)1024 * 512 * 2;    // 1 MB
  _Float16* Wihh = (_Float16*)w; w += (size_t)1024 * 1024 * 2;   // 2 MB
  _Float16* Whhh = (_Float16*)w; w += (size_t)1024 * 256 * 2;    // 0.5 MB
  _Float16* H0h  = (_Float16*)w; w += (size_t)512 * 256 * 2;     // 256 KB
  unsigned int* bar = (unsigned int*)w; w += 256;
  _Float16* H1h  = (_Float16*)w; w += (size_t)NB_ * 1024 * 2;    // 256 MB
  _Float16* HSh  = (_Float16*)w; w += (size_t)NB_ * 256 * 2;     // 64 MB
  float*    GX   = (float*)w;                                    // 512 MB

  k_zero<<<1, 32, 0, stream>>>(bar);
  k_cvt2d<<<dim3((1024 * 512 + 255) / 256),  256, 0, stream>>>(W1,   W1h,  1024, 512,  512);
  k_cvt2d<<<dim3((1024 * 1024 + 255) / 256), 256, 0, stream>>>(W_ih, Wihh, 1024, 1024, 1041);
  k_cvt2d<<<dim3((1024 * 256 + 255) / 256),  256, 0, stream>>>(W_hh, Whhh, 1024, 256,  256);
  k_cvt2d<<<dim3((512 * 256 + 255) / 256),   256, 0, stream>>>(hx,   H0h,  512,  256,  256);

  k_gemm1<<<dim3(NB_ / 128, HID_ / 128), 256, 0, stream>>>(x, W1h, b1, H1h);
  k_gemm2<<<dim3(NB_ / 128, HID_ / 128), 256, 0, stream>>>(H1h, Wihh, W_ih, b_ih, b_hh,
                                                           reward, act, GX);
  k_lstm<<<dim3(16), 512, 0, stream>>>(GX, Whhh, H0h, hx, dones, HSh, bar);
  k_heads<<<dim3(NB_ / 8), 160, 0, stream>>>(HSh, W_pi, b_pi, W_v, b_v, out);
}
